// RouterOursSoftmaxAddAttentionGatingNoNewToken_32830730011542
// MI455X (gfx1250) — matrix operand order repaired and checked by reading, not compiled, not run on hardware
//
#include <hip/hip_runtime.h>
#include <hip/hip_bf16.h>

typedef __attribute__((ext_vector_type(16))) _Float16 v16h;
typedef __attribute__((ext_vector_type(8)))  _Float16 v8h;
typedef __attribute__((ext_vector_type(4)))  _Float16 v4h;
typedef __attribute__((ext_vector_type(8)))  float    v8f;
typedef __attribute__((ext_vector_type(4)))  float    f4;
typedef __attribute__((ext_vector_type(4)))  unsigned int u4;

#define LN_EPS 1e-5f
// B=4, H=16, L=2048, D=1024 ; rows = B*L = 8192 ; score rows = B*H*L = 131072

// ---------------------------------------------------------------- zero scratch
__global__ void k_zero(float* __restrict__ p, int n) {
    int i = blockIdx.x * 256 + threadIdx.x;
    if (i < n) p[i] = 0.0f;
}

// --------------------------------------------- streaming score sum over (h, q)
// scores: [B,H,L,L] fp32 (1.07 GB).  2048 blocks x 256 threads; each block folds
// 64 rows of length 2048 into registers (8 accumulators/thread), then atomics.
__global__ void __launch_bounds__(256) k_score_sum(const float* __restrict__ scores,
                                                   float* __restrict__ accum) {
    const int t = threadIdx.x;
    const long long row0 = (long long)blockIdx.x * 64;      // flattened (b,h,q)
    const int b = (int)(row0 >> 15);                        // / (H*L)=32768
    f4 a0 = {0.f,0.f,0.f,0.f}, a1 = {0.f,0.f,0.f,0.f};
    const float* p = scores + row0 * 2048 + 4 * t;
    #pragma unroll 4
    for (int r = 0; r < 64; ++r) {
        f4 x0 = __builtin_nontemporal_load((const f4*)p);
        f4 x1 = __builtin_nontemporal_load((const f4*)(p + 1024));
        a0 += x0; a1 += x1;
        p += 2048;
    }
    float* d0 = accum + b * 2048 + 4 * t;
    #pragma unroll
    for (int c = 0; c < 4; ++c) {
        atomicAdd(d0 + c,        a0[c]);
        atomicAdd(d0 + 1024 + c, a1[c]);
    }
}

// ------------------------------------------------- importance normalize per b
__global__ void k_imp(const float* __restrict__ accum, float* __restrict__ imp) {
    const int b = blockIdx.x, t = threadIdx.x;
    __shared__ float smn[256], smx[256];
    float mn = 3.402823466e38f, mx = -3.402823466e38f;
    float v[8];
    #pragma unroll
    for (int i = 0; i < 8; ++i) {
        int k = t + i * 256;
        float s = accum[b * 2048 + k] * (1.0f / 32768.0f);  // mean over H*L
        v[i] = s;
        if (k >= 1) { mn = fminf(mn, s); mx = fmaxf(mx, s); }
    }
    smn[t] = mn; smx[t] = mx; __syncthreads();
    for (int s = 128; s > 0; s >>= 1) {
        if (t < s) { smn[t] = fminf(smn[t], smn[t+s]); smx[t] = fmaxf(smx[t], smx[t+s]); }
        __syncthreads();
    }
    mn = smn[0]; mx = smx[0];
    #pragma unroll
    for (int i = 0; i < 8; ++i) {
        int k = t + i * 256;
        imp[b * 2048 + k] = (k == 0) ? 1.0f : (v[i] - mn) / mx;  // faithful: /max
    }
}

// ---------------------------------------------------- LN1 -> f16 (GEMM input)
__global__ void __launch_bounds__(256) k_ln1(const float* __restrict__ x,
                                             const float* __restrict__ g,
                                             const float* __restrict__ bb,
                                             _Float16* __restrict__ y) {
    const int row = blockIdx.x, t = threadIdx.x;
    const float* xr = x + (long long)row * 1024;
    f4 xv = *(const f4*)(xr + 4 * t);
    float s1 = xv[0]+xv[1]+xv[2]+xv[3];
    float s2 = xv[0]*xv[0]+xv[1]*xv[1]+xv[2]*xv[2]+xv[3]*xv[3];
    __shared__ float r1[256], r2[256];
    r1[t] = s1; r2[t] = s2; __syncthreads();
    for (int s = 128; s > 0; s >>= 1) {
        if (t < s) { r1[t] += r1[t+s]; r2[t] += r2[t+s]; }
        __syncthreads();
    }
    float m   = r1[0] * (1.0f / 1024.0f);
    float var = r2[0] * (1.0f / 1024.0f) - m * m;
    float rstd = rsqrtf(var + LN_EPS);
    f4 gv = *(const f4*)(g + 4 * t);
    f4 bv = *(const f4*)(bb + 4 * t);
    v4h out;
    #pragma unroll
    for (int c = 0; c < 4; ++c)
        out[c] = (_Float16)((xv[c] - m) * rstd * gv[c] + bv[c]);
    *(v4h*)(y + (long long)row * 1024 + 4 * t) = out;
}

// --------------------------------------- W1 [K,N] -> W1^T [N,K] f16 (for WMMA B)
__global__ void k_w1t(const float* __restrict__ W1, _Float16* __restrict__ W1t) {
    __shared__ float tile[32][33];
    const int tx = threadIdx.x & 31, ty = threadIdx.x >> 5;   // 32 x 8
    const int d0 = blockIdx.y * 32, e0 = blockIdx.x * 32;
    #pragma unroll
    for (int i = 0; i < 4; ++i) {
        int r = ty + i * 8;
        tile[r][tx] = W1[(d0 + r) * 1024 + e0 + tx];
    }
    __syncthreads();
    #pragma unroll
    for (int i = 0; i < 4; ++i) {
        int r = ty + i * 8;
        W1t[(e0 + r) * 1024 + d0 + tx] = (_Float16)tile[tx][r];
    }
}

// --------------------------------------------------------------- WMMA GEMM
// C[8192,1024] = h1f16[8192,1024] @ W1  (+ b1), Bt = W1^T row-major [N][K].
// Block: 256 thr (8 waves), tile 128x64, BK=32; wave grid 4x2, 32x32 per wave
// = 4 x v_wmma_f32_16x16x32_f16 per K-step.
__global__ void __launch_bounds__(256) k_gemm(const _Float16* __restrict__ A,
                                              const _Float16* __restrict__ Bt,
                                              const float* __restrict__ b1,
                                              float* __restrict__ C) {
    constexpr int BM = 128, BN = 64, BK = 32, LDT = 40;  // pad 8 halves: 16B align kept
    __shared__ _Float16 sA[BM * LDT];
    __shared__ _Float16 sB[BN * LDT];
    const int t = threadIdx.x;
    const int m_base = blockIdx.y * BM;
    const int n_base = blockIdx.x * BN;
    const int w = t >> 5, lane = t & 31;
    const int wm = w >> 1, wn = w & 1;          // 4 x 2 wave grid
    const int half = lane >> 4, r16 = lane & 15;

    v8f acc[2][2] = {};

    for (int kt = 0; kt < 1024; kt += BK) {
        // stage A: 128x32 halves = 512 x16B, 2 chunks/thread
        #pragma unroll
        for (int i = 0; i < 2; ++i) {
            int idx = t + i * 256;
            int row = idx >> 2, cg = idx & 3;
            *(u4*)(&sA[row * LDT + cg * 8]) =
                *(const u4*)(A + (long long)(m_base + row) * 1024 + kt + cg * 8);
        }
        // stage Bt: 64x32 halves = 256 x16B, 1 chunk/thread
        {
            int row = t >> 2, cg = t & 3;
            *(u4*)(&sB[row * LDT + cg * 8]) =
                *(const u4*)(Bt + (long long)(n_base + row) * 1024 + kt + cg * 8);
        }
        __syncthreads();

        #pragma unroll
        for (int mi = 0; mi < 2; ++mi) {
            // A frag 16x32: lane-half 0 holds K 0-7 & 16-23, half 1 holds 8-15 & 24-31
            const _Float16* pa = &sA[(wm * 32 + mi * 16 + r16) * LDT];
            v8h alo = *(const v8h*)(pa + half * 8);
            v8h ahi = *(const v8h*)(pa + 16 + half * 8);
            v16h afrag = __builtin_shufflevector(alo, ahi,
                0,1,2,3,4,5,6,7,8,9,10,11,12,13,14,15);
            #pragma unroll
            for (int ni = 0; ni < 2; ++ni) {
                // B frag 32x16: lane = column, lane-half 0 holds K 0-7 & 16-23, half 1 holds 8-15 & 24-31
                const _Float16* pb = &sB[(wn * 32 + ni * 16 + r16) * LDT];
                v8h blo = *(const v8h*)(pb + half * 8);
                v8h bhi = *(const v8h*)(pb + 16 + half * 8);
                v16h bfrag = __builtin_shufflevector(blo, bhi,
                    0,1,2,3,4,5,6,7,8,9,10,11,12,13,14,15);
                acc[mi][ni] = __builtin_amdgcn_wmma_f32_16x16x32_f16(
                    false, afrag, false, bfrag, (short)0, acc[mi][ni], false, false);
            }
        }
        __syncthreads();
    }

    // epilogue: D layout — n = tile_n + (lane&15), m = tile_m + half*8 + vgpr
    #pragma unroll
    for (int mi = 0; mi < 2; ++mi) {
        #pragma unroll
        for (int ni = 0; ni < 2; ++ni) {
            int n  = n_base + wn * 32 + ni * 16 + r16;
            int mb = m_base + wm * 32 + mi * 16 + half * 8;
            float bias = b1[n];
            #pragma unroll
            for (int j = 0; j < 8; ++j)
                C[(long long)(mb + j) * 1024 + n] = acc[mi][ni][j] + bias;
        }
    }
}

// ---------------------- LN2 + exact GELU + x@W2 + softmax + mask, per row
__global__ void __launch_bounds__(256) k_head(const float* __restrict__ h2,
                                              const float* __restrict__ g2,
                                              const float* __restrict__ bb2,
                                              const float* __restrict__ W2,
                                              const float* __restrict__ b2v,
                                              const float* __restrict__ imp,
                                              float* __restrict__ out2,
                                              float* __restrict__ out3) {
    const int row = blockIdx.x, t = threadIdx.x;
    const float* xr = h2 + (long long)row * 1024;
    f4 xv = *(const f4*)(xr + 4 * t);
    __shared__ float r1[256], r2[256];
    float s1 = xv[0]+xv[1]+xv[2]+xv[3];
    float s2 = xv[0]*xv[0]+xv[1]*xv[1]+xv[2]*xv[2]+xv[3]*xv[3];
    r1[t] = s1; r2[t] = s2; __syncthreads();
    for (int s = 128; s > 0; s >>= 1) {
        if (t < s) { r1[t] += r1[t+s]; r2[t] += r2[t+s]; }
        __syncthreads();
    }
    float m   = r1[0] * (1.0f / 1024.0f);
    float var = r2[0] * (1.0f / 1024.0f) - m * m;
    float rstd = rsqrtf(var + LN_EPS);
    __syncthreads();

    f4 gv = *(const f4*)(g2 + 4 * t);
    f4 bv = *(const f4*)(bb2 + 4 * t);
    float d0 = 0.f, d1 = 0.f;
    #pragma unroll
    for (int c = 0; c < 4; ++c) {
        float hn = (xv[c] - m) * rstd * gv[c] + bv[c];
        float ge = 0.5f * hn * (1.0f + erff(hn * 0.70710678118654752f));  // exact GELU
        int d = 4 * t + c;
        d0 += ge * W2[2 * d + 0];
        d1 += ge * W2[2 * d + 1];
    }
    r1[t] = d0; r2[t] = d1; __syncthreads();
    for (int s = 128; s > 0; s >>= 1) {
        if (t < s) { r1[t] += r1[t+s]; r2[t] += r2[t+s]; }
        __syncthreads();
    }
    if (t == 0) {
        int l = row & 2047;
        float z0 = r1[0] + b2v[0] + (l == 0 ? 100.0f : 0.0f);  // CLS keep bias
        float z1 = r2[0] + b2v[1];
        float mz = fmaxf(z0, z1);
        float e0 = __expf(z0 - mz), e1 = __expf(z1 - mz);
        float learned = e0 / (e0 + e1);
        float fin = 0.5f * (imp[row] + learned);
        float hard = (fin >= 0.5f) ? 1.0f : 0.0f;
        out3[row] = hard - fin + fin;   // straight-through (== hard in fwd)
        out2[row] = 1.0f;               // tome_out = ones
    }
}

// ----------------------------------------------------------------- launcher
extern "C" void kernel_launch(void* const* d_in, const int* in_sizes, int n_in,
                              void* d_out, int out_size, void* d_ws, size_t ws_size,
                              hipStream_t stream) {
    (void)in_sizes; (void)n_in; (void)out_size; (void)ws_size;
    const float* hidden = (const float*)d_in[0];    // [4,2048,1024]
    const float* amask  = (const float*)d_in[1];    // [4,1,1,2048]
    const float* scores = (const float*)d_in[2];    // [4,16,2048,2048]
    const float* ln1_g  = (const float*)d_in[5];
    const float* ln1_b  = (const float*)d_in[6];
    const float* W1     = (const float*)d_in[7];    // [1024,1024]
    const float* b1     = (const float*)d_in[8];
    const float* ln2_g  = (const float*)d_in[9];
    const float* ln2_b  = (const float*)d_in[10];
    const float* W2     = (const float*)d_in[11];   // [1024,2]
    const float* b2     = (const float*)d_in[12];

    char* ws = (char*)d_ws;
    float*    accum = (float*)(ws);                                   // 32 KB
    float*    imp   = (float*)(ws + 32768);                           // 32 KB
    _Float16* W1t   = (_Float16*)(ws + 65536);                        // 2 MB
    _Float16* h1    = (_Float16*)(ws + 65536 + 2097152);              // 16 MB
    float*    h2    = (float*)(ws + 65536 + 2097152 + 16777216);      // 33.5 MB

    float* out0 = (float*)d_out;           // hidden passthrough [8388608]
    float* out1 = out0 + 8388608;          // attention_mask     [8192]
    float* out2 = out1 + 8192;             // tome ones          [8192]
    float* out3 = out2 + 8192;             // learnable mask     [8192]

    hipMemcpyAsync(out0, hidden, sizeof(float) * 8388608, hipMemcpyDeviceToDevice, stream);
    hipMemcpyAsync(out1, amask,  sizeof(float) * 8192,    hipMemcpyDeviceToDevice, stream);

    k_zero<<<32, 256, 0, stream>>>(accum, 8192);
    k_score_sum<<<2048, 256, 0, stream>>>(scores, accum);
    k_imp<<<4, 256, 0, stream>>>(accum, imp);
    k_ln1<<<8192, 256, 0, stream>>>(hidden, ln1_g, ln1_b, h1);
    k_w1t<<<dim3(32, 32), 256, 0, stream>>>(W1, W1t);
    k_gemm<<<dim3(16, 64), 256, 0, stream>>>(h1, W1t, b1, h2);
    k_head<<<8192, 256, 0, stream>>>(h2, ln2_g, ln2_b, W2, b2, imp, out2, out3);
}
